// DNET_61881888800848
// MI455X (gfx1250) — matrix operand order repaired and checked by reading, not compiled, not run on hardware
//
#include <hip/hip_runtime.h>
#include <hip/hip_bf16.h>

// ---------------------------------------------------------------------------
// NConv depth-completion network for MI455X (gfx1250, wave32, WMMA).
//
// Every normalized conv = two implicit GEMMs sharing weights (numerator over
// x*c, denominator over c).  M = 8 out-channels padded to 16, K = Cin*ks*ks
// laid out as (ky*ks+kx)*Ctot + ch, N = pixels.  Workgroup = 4 output rows x
// 32 pixels (8 waves, each a 16-pixel WMMA tile).  Inputs staged to LDS as
// packed f16 pairs {x*c, c}; B fragments gathered via a precomputed per-k
// offset table (no div/mod in the hot loop); A fragments preloaded to VGPRs;
// f32 accumulation via v_wmma_f32_16x16x32_f16.  Nearest-2x upsampling for
// the decoder concat branches is fused into LDS staging.
// ---------------------------------------------------------------------------

typedef __attribute__((ext_vector_type(16))) _Float16 v16h;
typedef __attribute__((ext_vector_type(8)))  float    v8f;
typedef __attribute__((ext_vector_type(8)))  short    v8s;
typedef __attribute__((ext_vector_type(2)))  _Float16 h2;

#define NC_EPS 1e-20f

// ---------------- weight packing: OIHW -> [m(16) x Kpad] f16, K-major -------
// K index: k = (ky*ks + kx)*Ctot + ch  (Ctot = Cin, power of two)
__global__ void k_pack_w(const float* __restrict__ w, _Float16* __restrict__ dst,
                         int Ctot, int lg, int ks, int Kpad) {
    int i = blockIdx.x * blockDim.x + threadIdx.x;
    if (i >= 16 * Kpad) return;
    int m = i / Kpad;
    int k = i - m * Kpad;
    int ch = k & (Ctot - 1);
    int rk = k >> lg;                // = ky*ks + kx when in range
    float v = 0.f;
    if (m < 8 && rk < ks * ks) {
        int r  = rk / ks;
        int kx = rk - r * ks;
        v = w[(((size_t)m * Ctot + ch) * ks + r) * ks + kx];
    }
    dst[(size_t)m * Kpad + k] = (_Float16)v;
}

__global__ void k_wsum(const float* __restrict__ w, float* __restrict__ out, int n) {
    int o = threadIdx.x;
    if (o >= 8) return;
    float s = 0.f;
    for (int i = 0; i < n; ++i) s += w[(size_t)o * n + i];
    out[o] = s;
}

// ---------------- layer 1: nconv 1->8, 5x5, pad 2 (VALU; K is tiny) ---------
__global__ void k_nconv1(const float* __restrict__ S,
                         const float* __restrict__ w1, const float* __restrict__ b1,
                         const float* __restrict__ wsum1,
                         float* __restrict__ ox, float* __restrict__ oc,
                         int H, int W) {
    int x = blockIdx.x * blockDim.x + threadIdx.x;
    int y = blockIdx.y;
    int b = blockIdx.z;
    if (x >= W) return;
    const float* Sb = S + (size_t)b * H * W;
    float sv[25], cv[25];
#pragma unroll
    for (int ky = 0; ky < 5; ++ky)
#pragma unroll
        for (int kx = 0; kx < 5; ++kx) {
            int iy = y + ky - 2, ix = x + kx - 2;
            float s = 0.f;
            if (iy >= 0 && iy < H && ix >= 0 && ix < W) s = Sb[(size_t)iy * W + ix];
            float c = (s > 0.01f) ? 1.f : 0.f;
            sv[ky * 5 + kx] = s * c;
            cv[ky * 5 + kx] = c;
        }
#pragma unroll
    for (int o = 0; o < 8; ++o) {
        float n = 0.f, d = 0.f;
#pragma unroll
        for (int t = 0; t < 25; ++t) {
            float w = w1[o * 25 + t];
            n += w * sv[t];
            d += w * cv[t];
        }
        size_t oi = (((size_t)b * 8 + o) * H + y) * (size_t)W + x;
        ox[oi] = n / (d + NC_EPS) + b1[o];
        oc[oi] = d / wsum1[o];
    }
}

// ---------------- argmax-of-conf 2x2 pool (first-max tie-break), conf/4 -----
__global__ void k_pool(const float* __restrict__ ix, const float* __restrict__ ic,
                       float* __restrict__ ox, float* __restrict__ oc,
                       int Ho, int Wo) {
    int x = blockIdx.x * blockDim.x + threadIdx.x;
    if (x >= Wo) return;
    int y  = blockIdx.y;
    int bc = blockIdx.z;                       // fused (batch*8 + channel)
    int Wi = 2 * Wo;
    size_t base = ((size_t)bc * (2 * Ho) + 2 * y) * Wi + 2 * x;
    float c00 = ic[base],      c01 = ic[base + 1];
    float c10 = ic[base + Wi], c11 = ic[base + Wi + 1];
    float x00 = ix[base],      x01 = ix[base + 1];
    float x10 = ix[base + Wi], x11 = ix[base + Wi + 1];
    float bcv = c00, bxv = x00;                // window order (0,0)(0,1)(1,0)(1,1)
    if (c01 > bcv) { bcv = c01; bxv = x01; }
    if (c10 > bcv) { bcv = c10; bxv = x10; }
    if (c11 > bcv) { bcv = c11; bxv = x11; }
    size_t oi = ((size_t)bc * Ho + y) * (size_t)Wo + x;
    ox[oi] = bxv;
    oc[oi] = bcv * 0.25f;
}

// ---------------- generic WMMA nconv: Ctot in {8,16}, ks in {3,5} -----------
// Workgroup: 4 output rows x 32 pixels.  8 waves: wave w -> row (w>>1),
// 16-pixel segment ((w&1)*16).  group0 = channels 0..7, group1 = 8..15
// (nullptr if Ctot==8); upN==1 => group source at half resolution (fused up2).
template <int KS, int CTOT, int LG, int KPAD>
__global__ __launch_bounds__(256) void k_nconv_wmma(
    const float* __restrict__ x0, const float* __restrict__ c0, int up0,
    const float* __restrict__ x1, const float* __restrict__ c1, int up1,
    const _Float16* __restrict__ wp, const float* __restrict__ bias,
    const float* __restrict__ wsum,
    float* __restrict__ ox, float* __restrict__ oc,
    int Hin, int Win, int Ho, int Wo, int pad) {

    constexpr int PR   = KS + 3;               // patch rows (4 out rows + halo)
    constexpr int LW   = 31 + KS;              // patch cols (32 px + halo)
    constexpr int KS2  = KS * KS;
    constexpr int ELT  = CTOT * PR * LW;       // staged pairs
    constexpr int PADZ = 160;                  // zeroed pad region for k >= Kreal
    constexpr int NT   = KPAD >> 5;            // K tiles of 32

    __shared__ h2 sDC[ELT + PADZ];             // packed {f16(x*c), f16(c)}
    __shared__ alignas(32) short kOff[KPAD];   // k -> patch offset (or ELT)

    const int ywg = blockIdx.y * 4;
    const int b   = blockIdx.z;
    const int xwg = blockIdx.x * 32;
    const int tid = threadIdx.x;

    __builtin_prefetch(wp, 0, 1);              // global_prefetch_b8 (weights hot)

    // ---- per-k offset table: strips all div/mod from the WMMA loop ---------
    if (tid < KPAD) {
        int ch = tid & (CTOT - 1);
        int rk = tid >> LG;
        short off = (short)ELT;                // zero region
        if (rk < KS2) {
            int r  = rk / KS;
            int kx = rk - r * KS;
            off = (short)((ch * PR + r) * LW + kx);
        }
        kOff[tid] = off;
    }

    // ---- stage input patch as packed f16 pairs -----------------------------
    for (int i = tid; i < ELT + PADZ; i += 256) {
        float vx = 0.f, vc = 0.f;
        if (i < ELT) {
            int cc = i % LW;
            int t  = i / LW;
            int rr = t % PR;
            int ch = t / PR;
            int iy = ywg + rr - pad;
            int ix = xwg + cc - pad;
            if (iy >= 0 && iy < Hin && ix >= 0 && ix < Win) {
                int g   = ch >> 3;
                int icc = ch & 7;
                const float* px = g ? x1 : x0;
                const float* pc = g ? c1 : c0;
                int up = g ? up1 : up0;
                size_t si;
                if (up) si = (((size_t)b * 8 + icc) * (Hin >> 1) + (iy >> 1)) * (size_t)(Win >> 1) + (ix >> 1);
                else    si = (((size_t)b * 8 + icc) * Hin + iy) * (size_t)Win + ix;
                vc = pc[si];
                vx = px[si] * vc;
            }
        }
        h2 pr = {(_Float16)vx, (_Float16)vc};
        sDC[i] = pr;
    }
    __syncthreads();

    // ---- WMMA implicit GEMM over K tiles of 32 -----------------------------
    const int lane = tid & 31;
    const int wv   = tid >> 5;
    const int yr   = wv >> 1;                  // output row within block (0..3)
    const int xl   = (wv & 1) * 16 + (lane & 15);  // pixel within 32-px chunk
    const int col  = lane & 15;                // B-matrix column
    const int hiK  = (lane & 16) ? 1 : 0;      // lanes 16..31: K+16 (A and B)
    const int m    = lane & 15;                // A-matrix row (out channel)
    const int laneBase = yr * LW + xl;

    // A fragments preloaded (contiguous k order: v0..7 -> K0..15(+16))
    v16h af[NT];
#pragma unroll
    for (int t = 0; t < NT; ++t) {
        const int kt = t << 5;
#pragma unroll
        for (int v = 0; v < 8; ++v) {
            int kLo = 2 * v + (hiK ? 16 : 0);
            int ka  = kt + kLo;
            af[t][2 * v]     = wp[(size_t)m * KPAD + ka];
            af[t][2 * v + 1] = wp[(size_t)m * KPAD + ka + 1];
        }
    }

    v8f accN = {};                             // numerator  (x*c GEMM)
    v8f accD = {};                             // denominator (c GEMM)

#pragma unroll
    for (int t = 0; t < NT; ++t) {
        const int kt = t << 5;
        // 16 consecutive k-offsets for this lane: two aligned b128 LDS reads
        const v8s* kp = (const v8s*)&kOff[kt + (hiK << 4)];
        v8s k0 = kp[0], k1 = kp[1];
        v16h bn, bc;
#pragma unroll
        for (int e = 0; e < 16; ++e) {
            int off  = (e < 8) ? (int)k0[e] : (int)k1[e - 8];
            h2  q    = sDC[off + laneBase];    // one ds_load_b32: {num, den}
            bn[e] = q.x;
            bc[e] = q.y;
        }
        accN = __builtin_amdgcn_wmma_f32_16x16x32_f16(false, af[t], false, bn,
                                                      (short)0, accN, false, false);
        accD = __builtin_amdgcn_wmma_f32_16x16x32_f16(false, af[t], false, bc,
                                                      (short)0, accD, false, false);
    }

    // ---- epilogue: lanes 0..15 hold rows 0..7 (the 8 real out-channels) ----
    int x = xwg + xl;
    int y = ywg + yr;
    if (lane < 16 && x < Wo && y < Ho) {
#pragma unroll
        for (int r = 0; r < 8; ++r) {
            float den = accD[r];
            float xo  = accN[r] / (den + NC_EPS) + bias[r];
            float co  = den / wsum[r];
            size_t oi = (((size_t)b * 8 + r) * Ho + y) * (size_t)Wo + x;
            ox[oi] = xo;
            oc[oi] = co;
        }
    }
}

// ---------------- final: 1x1 nconv pad 2 + slice [1:481,1:641] --------------
__global__ void k_final(const float* __restrict__ xin, const float* __restrict__ cin,
                        const float* __restrict__ w7, const float* __restrict__ b7,
                        float* __restrict__ out, int B) {
    const int H = 480, W = 640, Hi = 478, Wi = 638;
    int x = blockIdx.x * blockDim.x + threadIdx.x;
    if (x >= W) return;
    int y = blockIdx.y;
    int b = blockIdx.z;
    int iy = y - 1, ix = x - 1;                // slice offset: padded row/col -1
    float xo, co;
    if (iy >= 0 && iy < Hi && ix >= 0 && ix < Wi) {
        float n = 0.f, d = 0.f, sw = 0.f;
#pragma unroll
        for (int icn = 0; icn < 8; ++icn) {
            float w = w7[icn];
            sw += w;
            size_t si = (((size_t)b * 8 + icn) * Hi + iy) * (size_t)Wi + ix;
            float cc = cin[si];
            float xx = xin[si];
            d += w * cc;
            n += w * xx * cc;
        }
        xo = n / (d + NC_EPS) + b7[0];
        co = d / sw;
    } else {                                   // pure zero-padding region
        xo = b7[0];
        co = 0.f;
    }
    size_t oi = ((size_t)b * H + y) * (size_t)W + x;
    out[oi] = xo;
    out[(size_t)B * H * W + oi] = co;
}

// ---------------------------------------------------------------------------
extern "C" void kernel_launch(void* const* d_in, const int* in_sizes, int n_in,
                              void* d_out, int out_size, void* d_ws, size_t ws_size,
                              hipStream_t stream) {
    (void)in_sizes; (void)n_in; (void)out_size; (void)ws_size;

    // setup_inputs order: fout,S,K,w1,b1,w2,b2,w3,b3,w4,b4,w5,b5,w6,b6,w7,b7
    const float* S  = (const float*)d_in[1];
    const float* w1 = (const float*)d_in[3];  const float* b1 = (const float*)d_in[4];
    const float* w2 = (const float*)d_in[5];  const float* b2 = (const float*)d_in[6];
    const float* w3 = (const float*)d_in[7];  const float* b3 = (const float*)d_in[8];
    const float* w4 = (const float*)d_in[9];  const float* b4 = (const float*)d_in[10];
    const float* w5 = (const float*)d_in[11]; const float* b5 = (const float*)d_in[12];
    const float* w6 = (const float*)d_in[13]; const float* b6 = (const float*)d_in[14];
    const float* w7 = (const float*)d_in[15]; const float* b7 = (const float*)d_in[16];

    // -------- workspace layout (bytes, 256-aligned), with buffer reuse ------
    char* ws = (char*)d_ws;
    size_t o = 0;
    auto A = [&](size_t n) { size_t r = o; o = (o + n + 255) & ~(size_t)255; return r; };
    const size_t WP2 = A(16 * 224 * 2), WP3 = A(16 * 224 * 2);
    const size_t WP4 = A(16 * 160 * 2), WP5 = A(16 * 160 * 2), WP6 = A(16 * 160 * 2);
    const size_t WS1 = A(32), WS2 = A(32), WS3 = A(32), WS4 = A(32), WS5 = A(32), WS6 = A(32);
    const size_t NF = 19660800ull;   // 8*8*480*640
    const size_t NH = 4915200ull;    // 8*8*240*320
    const size_t NQ = 1228800ull;    // 8*8*120*160
    const size_t NE = 307200ull;     // 8*8*60*80
    const size_t T1X = A(NF * 4), T1C = A(NF * 4);       // full (L1 out, later x1/c1)
    const size_t T2X = A(NF * 4), T2C = A(NF * 4);       // full scratch, later w6 out (478x638)
    const size_t P1X = A(NH * 4), P1C = A(NH * 4);       // pool1 out, later x23
    const size_t H1X = A(NH * 4), H1C = A(NH * 4);       // w2@half out
    const size_t X2X = A(NH * 4), X2C = A(NH * 4);       // x2_ds (kept)
    const size_t P2X = A(NQ * 4), P2C = A(NQ * 4);       // pool2 out, later x34
    const size_t X3X = A(NQ * 4), X3C = A(NQ * 4);       // x3_ds (kept)
    const size_t P3X = A(NE * 4), P3C = A(NE * 4);       // pool3 out
    const size_t X4X = A(NE * 4), X4C = A(NE * 4);       // x4_ds
    auto F = [&](size_t off) { return (float*)(ws + off); };
    auto Hh = [&](size_t off) { return (_Float16*)(ws + off); };

    // -------- weight prep ---------------------------------------------------
    k_wsum<<<1, 8, 0, stream>>>(w1, F(WS1), 25);
    k_wsum<<<1, 8, 0, stream>>>(w2, F(WS2), 200);
    k_wsum<<<1, 8, 0, stream>>>(w3, F(WS3), 200);
    k_wsum<<<1, 8, 0, stream>>>(w4, F(WS4), 144);
    k_wsum<<<1, 8, 0, stream>>>(w5, F(WS5), 144);
    k_wsum<<<1, 8, 0, stream>>>(w6, F(WS6), 144);
    k_pack_w<<<(16 * 224 + 255) / 256, 256, 0, stream>>>(w2, Hh(WP2), 8, 3, 5, 224);
    k_pack_w<<<(16 * 224 + 255) / 256, 256, 0, stream>>>(w3, Hh(WP3), 8, 3, 5, 224);
    k_pack_w<<<(16 * 160 + 255) / 256, 256, 0, stream>>>(w4, Hh(WP4), 16, 4, 3, 160);
    k_pack_w<<<(16 * 160 + 255) / 256, 256, 0, stream>>>(w5, Hh(WP5), 16, 4, 3, 160);
    k_pack_w<<<(16 * 160 + 255) / 256, 256, 0, stream>>>(w6, Hh(WP6), 16, 4, 3, 160);

    // -------- encoder -------------------------------------------------------
    // L1: nconv(S, c0, w1) -> T1 [8,8,480,640]
    k_nconv1<<<dim3(3, 480, 8), 256, 0, stream>>>(S, w1, b1, F(WS1), F(T1X), F(T1C), 480, 640);
    // L2: w2 @ full -> T2
    k_nconv_wmma<5, 8, 3, 224><<<dim3(20, 120, 8), 256, 0, stream>>>(
        F(T1X), F(T1C), 0, nullptr, nullptr, 0, Hh(WP2), b2, F(WS2),
        F(T2X), F(T2C), 480, 640, 480, 640, 2);
    // L3: w3 @ full -> T1 (= x1,c1, kept for w6 concat)
    k_nconv_wmma<5, 8, 3, 224><<<dim3(20, 120, 8), 256, 0, stream>>>(
        F(T2X), F(T2C), 0, nullptr, nullptr, 0, Hh(WP3), b3, F(WS3),
        F(T1X), F(T1C), 480, 640, 480, 640, 2);
    // pool1 -> P1 [240x320]
    k_pool<<<dim3(2, 240, 64), 256, 0, stream>>>(F(T1X), F(T1C), F(P1X), F(P1C), 240, 320);
    // w2 @ half -> H1
    k_nconv_wmma<5, 8, 3, 224><<<dim3(10, 60, 8), 256, 0, stream>>>(
        F(P1X), F(P1C), 0, nullptr, nullptr, 0, Hh(WP2), b2, F(WS2),
        F(H1X), F(H1C), 240, 320, 240, 320, 2);
    // w3 @ half -> X2 (= x2_ds, kept)
    k_nconv_wmma<5, 8, 3, 224><<<dim3(10, 60, 8), 256, 0, stream>>>(
        F(H1X), F(H1C), 0, nullptr, nullptr, 0, Hh(WP3), b3, F(WS3),
        F(X2X), F(X2C), 240, 320, 240, 320, 2);
    // pool2 -> P2 [120x160]
    k_pool<<<dim3(1, 120, 64), 256, 0, stream>>>(F(X2X), F(X2C), F(P2X), F(P2C), 120, 160);
    // w2 @ quarter -> X3 (= x3_ds, kept)
    k_nconv_wmma<5, 8, 3, 224><<<dim3(5, 30, 8), 256, 0, stream>>>(
        F(P2X), F(P2C), 0, nullptr, nullptr, 0, Hh(WP2), b2, F(WS2),
        F(X3X), F(X3C), 120, 160, 120, 160, 2);
    // pool3 -> P3 [60x80]
    k_pool<<<dim3(1, 60, 64), 256, 0, stream>>>(F(X3X), F(X3C), F(P3X), F(P3C), 60, 80);
    // w2 @ eighth -> X4 (= x4_ds)
    k_nconv_wmma<5, 8, 3, 224><<<dim3(3, 15, 8), 256, 0, stream>>>(
        F(P3X), F(P3C), 0, nullptr, nullptr, 0, Hh(WP2), b2, F(WS2),
        F(X4X), F(X4C), 60, 80, 60, 80, 2);

    // -------- decoder (concat convs, upsample fused via up flag) ------------
    // w4: cat(x3_ds, up(x4_ds)) @ quarter -> P2 (= x34_ds)
    k_nconv_wmma<3, 16, 4, 160><<<dim3(5, 30, 8), 256, 0, stream>>>(
        F(X3X), F(X3C), 0, F(X4X), F(X4C), 1, Hh(WP4), b4, F(WS4),
        F(P2X), F(P2C), 120, 160, 120, 160, 1);
    // w5: cat(x2_ds, up(x34_ds)) @ half -> P1 (= x23_ds)
    k_nconv_wmma<3, 16, 4, 160><<<dim3(10, 60, 8), 256, 0, stream>>>(
        F(X2X), F(X2C), 0, F(P2X), F(P2C), 1, Hh(WP5), b5, F(WS5),
        F(P1X), F(P1C), 240, 320, 240, 320, 1);
    // w6: cat(up(x23_ds), x1) @ full, pad 0 -> T2 [8,8,478,638]
    k_nconv_wmma<3, 16, 4, 160><<<dim3(20, 120, 8), 256, 0, stream>>>(
        F(P1X), F(P1C), 1, F(T1X), F(T1C), 0, Hh(WP6), b6, F(WS6),
        F(T2X), F(T2C), 480, 640, 478, 638, 0);

    // -------- final 1x1 nconv (pad 2) + slice -> d_out ----------------------
    k_final<<<dim3(3, 480, 8), 256, 0, stream>>>(F(T2X), F(T2C), w7, b7, (float*)d_out, 8);
}
